// TransformerTaskHead_Autoregressive_66383014527664
// MI455X (gfx1250) — matrix operand order repaired and checked by reading, not compiled, not run on hardware
//
#include <hip/hip_runtime.h>
#include <hip/hip_bf16.h>

// ---------------------------------------------------------------------------
// CDNA5 / gfx1250 transformer decoder.
//  - All GEMM-like math on v_wmma_f32_16x16x32_bf16 (wave32, 8xVGPR f32 acc).
//  - GEMM tile staging via the Tensor Data Mover (tensor_load_to_lds),
//    double-buffered on TENSORcnt, with D# pad fields reproducing the padded
//    LDS row stride.  Falls back to manual b128 staging if the builtin is
//    unavailable.
// ---------------------------------------------------------------------------

typedef __bf16 bf16;
typedef __attribute__((ext_vector_type(16))) __bf16 v16bf;
typedef __attribute__((ext_vector_type(8)))  float  v8f;
typedef __attribute__((ext_vector_type(4)))  unsigned int v4u;
typedef __attribute__((ext_vector_type(8)))  int v8i;
typedef __attribute__((ext_vector_type(4)))  int v4i;

#if defined(__has_builtin)
#if __has_builtin(__builtin_amdgcn_tensor_load_to_lds) && \
    __has_builtin(__builtin_amdgcn_s_wait_tensorcnt)
#define USE_TDM 1
#endif
#endif
#ifndef USE_TDM
#define USE_TDM 0
#endif

union FragU { unsigned int u[8]; v16bf v; };
union BFU   { unsigned short s; bf16 h; };

__device__ __forceinline__ v8f wmma_bf16f32(v16bf a, v16bf b, v8f c) {
  // 8 args: (neg_a, A, neg_b, B, c_mod, C, reuse_a, reuse_b)
  return __builtin_amdgcn_wmma_f32_16x16x32_bf16(false, a, false, b, (short)0, c,
                                                 false, false);
}

__device__ __forceinline__ unsigned int bf16bits(bf16 h) {
  BFU u; u.h = h; return (unsigned int)u.s;
}

// ---------------------------------------------------------------------------
// f32 -> bf16 convert (grid-stride)
// ---------------------------------------------------------------------------
__global__ __launch_bounds__(256) void f32_to_bf16_kernel(
    const float* __restrict__ in, bf16* __restrict__ out, long long n) {
  long long i = (long long)blockIdx.x * blockDim.x + threadIdx.x;
  long long stride = (long long)gridDim.x * blockDim.x;
  for (; i < n; i += stride) out[i] = (bf16)in[i];
}

// ---------------------------------------------------------------------------
// Embedding + positional embedding.  rows = B*T (2016), D=512.
// ---------------------------------------------------------------------------
__global__ __launch_bounds__(256) void embed_kernel(
    const int* __restrict__ ids, const float* __restrict__ emb,
    const float* __restrict__ pos, float* __restrict__ x, bf16* __restrict__ xb) {
  const int row = blockIdx.x;          // b*63 + t
  const int b = row / 63, t = row % 63;
  const int tok = ids[b * 64 + t];     // tgt = input_ids[:, :-1]
  const float* e = emb + (size_t)tok * 512;
  const float* p = pos + (size_t)t * 512;
  for (int i = threadIdx.x; i < 512; i += 256) {
    float v = e[i] + p[i];
    x[(size_t)row * 512 + i]  = v;
    xb[(size_t)row * 512 + i] = (bf16)v;
  }
}

// ---------------------------------------------------------------------------
// residual add + LayerNorm over D=512, one row per block (256 thr x 2 elems)
// ---------------------------------------------------------------------------
__global__ __launch_bounds__(256) void add_ln_kernel(
    const float* __restrict__ xin, const float* __restrict__ h,
    const float* __restrict__ g, const float* __restrict__ bb,
    float* __restrict__ xo, bf16* __restrict__ xob) {
  __shared__ float red[256];
  const int row = blockIdx.x;
  const int tid = threadIdx.x;
  const float* xr = xin + (size_t)row * 512;
  const float* hr = h   + (size_t)row * 512;
  float v0 = xr[tid]       + hr[tid];
  float v1 = xr[tid + 256] + hr[tid + 256];
  red[tid] = v0 + v1;
  __syncthreads();
  for (int s = 128; s > 0; s >>= 1) {
    if (tid < s) red[tid] += red[tid + s];
    __syncthreads();
  }
  const float mu = red[0] * (1.f / 512.f);
  __syncthreads();
  const float d0 = v0 - mu, d1 = v1 - mu;
  red[tid] = d0 * d0 + d1 * d1;
  __syncthreads();
  for (int s = 128; s > 0; s >>= 1) {
    if (tid < s) red[tid] += red[tid + s];
    __syncthreads();
  }
  const float rstd = rsqrtf(red[0] * (1.f / 512.f) + 1e-5f);
  const float o0 = d0 * rstd * g[tid]       + bb[tid];
  const float o1 = d1 * rstd * g[tid + 256] + bb[tid + 256];
  xo[(size_t)row * 512 + tid]        = o0;
  xo[(size_t)row * 512 + tid + 256]  = o1;
  xob[(size_t)row * 512 + tid]       = (bf16)o0;
  xob[(size_t)row * 512 + tid + 256] = (bf16)o1;
}

// ---------------------------------------------------------------------------
// Tiled bf16 WMMA GEMM:  C[M,N] = A[M,K] @ W[N,K]^T + bias, optional ReLU.
// Block tile 128x128x32, 8 waves: 4(M) x 2(N), each wave 32x64 = 8 WMMA/step.
// Staging: TDM tensor_load_to_lds, double-buffered on TENSORcnt.
// ---------------------------------------------------------------------------
#define GBM 128
#define GBN 128
#define GBK 32
#define GLP 40                       // padded LDS row stride (bf16): 64B+16B pad
#define GTILE_BYTES (GBM * GLP * 2)  // 10240 bytes per staged tile

#if USE_TDM
// 2D tile load via Tensor Data Mover.  D# fields per CDNA5 ISA ch.8:
//  group0: count=1 | lds_addr | global_addr(57b) | type=2
//  group1: data_size=2B, pad_enable, pad_interval=3 (16 DWORDs = one 64B row),
//          pad_amount=3 (4 DWORDs = 16B) -> LDS row stride 80B == GLP*2.
//          tensor_dim0=K, tensor_dim1=rows remaining, tile 32(K) x 128(rows),
//          dim0 stride = K elements.
__device__ __forceinline__ void tdm_load_tile(unsigned lds_off,
                                              const bf16* g, int Kelems,
                                              int rows) {
  const unsigned long long ga = (unsigned long long)(uintptr_t)g;
  v4u g0;
  g0[0] = 1u;                                  // count=1, user descriptor
  g0[1] = lds_off;                             // lds_addr (bytes)
  g0[2] = (unsigned)ga;                        // global_addr[31:0]
  g0[3] = (unsigned)((ga >> 32) & 0x01ffffffu) // global_addr[56:32]
          | (2u << 30);                        // type=2 ("image")
  v8i g1;
  g1[0] = (int)((1u << 16)                     // data_size = 2 bytes
              | (1u << 20)                     // pad_enable
              | (3u << 22)                     // pad_interval: 16 DWORDs
              | (3u << 25));                   // pad_amount:   4 DWORDs
  g1[1] = (int)(((unsigned)Kelems & 0xffffu) << 16);          // tensor_dim0 lo
  g1[2] = (int)(((unsigned)Kelems >> 16) |
                (((unsigned)rows & 0xffffu) << 16));          // td0 hi | td1 lo
  g1[3] = (int)((((unsigned)rows >> 16) & 0xffffu) |
                ((unsigned)GBK << 16));                       // td1 hi | tile0
  g1[4] = (int)(GBM & 0xffff);                                // tile1, tile2=0
  g1[5] = Kelems;                                             // dim0 stride lo
  g1[6] = 0;                                                  // stride hi
  g1[7] = 0;
  const v4i z4 = {0, 0, 0, 0};
  const v8i z8 = {0, 0, 0, 0, 0, 0, 0, 0};
  // 6-arg form (clang-23 / therock-10.0): (g0, g1, g2, g3, g4, cpol)
  __builtin_amdgcn_tensor_load_to_lds(g0, g1, z4, z4, z8, 0);
}
#endif

__device__ __forceinline__ void gemm_step(const bf16* __restrict__ Al,
                                          const bf16* __restrict__ Bl,
                                          v8f (&acc)[2][4], int wm, int wn,
                                          int l15, int khalf, int kb) {
  v16bf afrag[2];
#pragma unroll
  for (int mt = 0; mt < 2; ++mt) {
    FragU fa;
    const bf16* ap = &Al[(wm + mt * 16 + l15) * GLP];
#pragma unroll
    for (int d = 0; d < 8; ++d) {
      const int klo = (d < 4 ? 2 * d : 16 + 2 * (d - 4)) + khalf;
      fa.u[d] = *(const unsigned int*)(ap + klo);
    }
    afrag[mt] = fa.v;
  }
#pragma unroll
  for (int nt = 0; nt < 4; ++nt) {
    FragU fb;
    const bf16* bp = &Bl[(wn + nt * 16 + l15) * GLP];
#pragma unroll
    for (int d = 0; d < 8; ++d)
      fb.u[d] = *(const unsigned int*)(bp + (d < 4 ? 2 * d : 16 + 2 * (d - 4)) + khalf);
#pragma unroll
    for (int mt = 0; mt < 2; ++mt)
      acc[mt][nt] = wmma_bf16f32(afrag[mt], fb.v, acc[mt][nt]);
  }
}

__global__ __launch_bounds__(256) void gemm_bf16_kernel(
    const bf16* __restrict__ A, const bf16* __restrict__ W,
    const float* __restrict__ bias,
    float* __restrict__ Cf, bf16* __restrict__ Cb,
    int M, int N, int K, int relu) {
  extern __shared__ char smem[];  // 2 buffers x (A tile + W tile)

  const int tid  = threadIdx.x;
  const int lane = tid & 31;
  const int wv   = tid >> 5;
  const int bm = blockIdx.y * GBM;
  const int bn = blockIdx.x * GBN;
  const int wm = (wv & 3) * 32;
  const int wn = (wv >> 2) * 64;
  const int l15   = lane & 15;
  const int mh    = (lane >> 4) * 8;
  const int khalf = (lane >> 4) * 8;
  const int kb    = (lane >> 4) * 16;

  v8f acc[2][4];
#pragma unroll
  for (int i = 0; i < 2; ++i)
#pragma unroll
    for (int j = 0; j < 4; ++j)
#pragma unroll
      for (int r = 0; r < 8; ++r) acc[i][j][r] = 0.f;

#if USE_TDM
  // wave-uniform predicate so the issuing region is a scalar branch
  // (TDM ignores EXEC, so an exec-masked region would multi-issue).
  const bool issuer = (__builtin_amdgcn_readfirstlane(wv) == 0);
  const int nk = K / GBK;
  if (issuer) {
    tdm_load_tile(0u,                   A + (size_t)bm * K, K, M - bm);
    tdm_load_tile((unsigned)GTILE_BYTES, W + (size_t)bn * K, K, N - bn);
  }
  for (int it = 0; it < nk; ++it) {
    const int p = it & 1;
    if (issuer) {
      if (it + 1 < nk) {
        const size_t kk2 = (size_t)(it + 1) * GBK;
        const unsigned dst = (unsigned)(1 - p) * 2u * (unsigned)GTILE_BYTES;
        tdm_load_tile(dst,                          A + (size_t)bm * K + kk2,
                      K, M - bm);
        tdm_load_tile(dst + (unsigned)GTILE_BYTES,  W + (size_t)bn * K + kk2,
                      K, N - bn);
        // two newest loads may stay in flight; current pair must be done
        __builtin_amdgcn_s_wait_tensorcnt((short)2);
      } else {
        __builtin_amdgcn_s_wait_tensorcnt((short)0);
      }
    }
    __syncthreads();  // publish TDM-filled buffer p to all waves
    const bf16* Al = (const bf16*)(smem + p * 2 * GTILE_BYTES);
    const bf16* Bl = (const bf16*)(smem + p * 2 * GTILE_BYTES + GTILE_BYTES);
    gemm_step(Al, Bl, acc, wm, wn, l15, khalf, kb);
    __syncthreads();  // reads done before TDM refills this buffer next round
  }
#else
  // fallback: manual b128 staging, single buffer
  bf16* AlW = (bf16*)(smem);
  bf16* BlW = (bf16*)(smem + GTILE_BYTES);
  const int lrow = tid >> 1;
  const int lcol = (tid & 1) * 16;
  for (int kk = 0; kk < K; kk += GBK) {
    {
      bf16* dst = &AlW[lrow * GLP + lcol];
      if (bm + lrow < M) {
        const bf16* src = A + (size_t)(bm + lrow) * K + kk + lcol;
        *(uint4*)(dst)     = *(const uint4*)(src);
        *(uint4*)(dst + 8) = *(const uint4*)(src + 8);
      } else {
        uint4 z; z.x = z.y = z.z = z.w = 0u;
        *(uint4*)(dst) = z; *(uint4*)(dst + 8) = z;
      }
    }
    {
      bf16* dst = &BlW[lrow * GLP + lcol];
      if (bn + lrow < N) {
        const bf16* src = W + (size_t)(bn + lrow) * K + kk + lcol;
        *(uint4*)(dst)     = *(const uint4*)(src);
        *(uint4*)(dst + 8) = *(const uint4*)(src + 8);
      } else {
        uint4 z; z.x = z.y = z.z = z.w = 0u;
        *(uint4*)(dst) = z; *(uint4*)(dst + 8) = z;
      }
    }
    __syncthreads();
    gemm_step(AlW, BlW, acc, wm, wn, l15, khalf, kb);
    __syncthreads();
  }
#endif

  // epilogue: bias, optional relu, f32 and/or bf16 stores
#pragma unroll
  for (int mt = 0; mt < 2; ++mt)
#pragma unroll
    for (int nt = 0; nt < 4; ++nt)
#pragma unroll
      for (int r = 0; r < 8; ++r) {
        const int m = bm + wm + mt * 16 + mh + r;
        const int n = bn + wn + nt * 16 + l15;
        if (m < M && n < N) {
          float v = acc[mt][nt][r];
          if (bias) v += bias[n];
          if (relu) v = fmaxf(v, 0.f);
          if (Cf) Cf[(size_t)m * N + n] = v;
          if (Cb) Cb[(size_t)m * N + n] = (bf16)v;
        }
      }
}

// ---------------------------------------------------------------------------
// Fused attention for one (batch, head): scores = softmax(mask(QK^T/8)) @ V.
// Tq=63 (padded to 64), Tk=63 (self, causal+pad) or 64 (cross, key padding),
// hd=64.  Both einsums run on WMMA; softmax on VALU over LDS scores.
// ---------------------------------------------------------------------------
#define ALP 72  // bf16 LDS stride
#define SLP 68  // f32  LDS stride

__global__ __launch_bounds__(256) void attention_kernel(
    const float* __restrict__ Qb, int qld, int qc0,
    const float* __restrict__ Kb, int kld, int kc0,
    const float* __restrict__ Vb, int vc0,
    int Tk, int causal,
    const int* __restrict__ ids,            // self: pad keys where token==0
    const unsigned char* __restrict__ kpad, // cross: memory_key_padding_mask
    bf16* __restrict__ O) {
  __shared__ bf16 Qs[64 * ALP];   // Q tile, later reused for attn probs
  __shared__ bf16 Ks[64 * ALP];
  __shared__ bf16 Vs[64 * ALP];
  __shared__ float Sc[64 * SLP];

  const int b = blockIdx.x >> 3;
  const int h = blockIdx.x & 7;
  const int tid  = threadIdx.x;
  const int lane = tid & 31;
  const int wv   = tid >> 5;

  {  // stage q/k/v head slices (f32 -> bf16), zero-pad rows
    const int r  = tid >> 2;          // 0..63
    const int c0 = (tid & 3) * 16;    // 0,16,32,48
    const float* qs = Qb + (size_t)(b * 63 + r) * qld + qc0 + h * 64 + c0;
    const float* ks = Kb + (size_t)(b * Tk + r) * kld + kc0 + h * 64 + c0;
    const float* vs = Vb + (size_t)(b * Tk + r) * kld + vc0 + h * 64 + c0;
#pragma unroll
    for (int j = 0; j < 16; ++j) {
      Qs[r * ALP + c0 + j] = (r < 63) ? (bf16)qs[j] : (bf16)0.f;
      Ks[r * ALP + c0 + j] = (r < Tk) ? (bf16)ks[j] : (bf16)0.f;
      Vs[r * ALP + c0 + j] = (r < Tk) ? (bf16)vs[j] : (bf16)0.f;
    }
  }
  __syncthreads();

  const int m0 = (wv & 3) * 16;
  const int n0 = (wv >> 2) * 32;
  const int l15   = lane & 15;
  const int mh    = (lane >> 4) * 8;
  const int khalf = (lane >> 4) * 8;
  const int kb    = (lane >> 4) * 16;

  // ---- phase A: S = Q @ K^T  (K dim = hd = 64 -> two k-steps) ----
  v8f sacc[2];
#pragma unroll
  for (int nt = 0; nt < 2; ++nt)
#pragma unroll
    for (int r = 0; r < 8; ++r) sacc[nt][r] = 0.f;

  for (int ks = 0; ks < 64; ks += 32) {
    FragU fa;
    const bf16* ap = &Qs[(m0 + l15) * ALP + ks];
#pragma unroll
    for (int d = 0; d < 8; ++d) {
      const int klo = (d < 4 ? 2 * d : 16 + 2 * (d - 4)) + khalf;
      fa.u[d] = *(const unsigned int*)(ap + klo);
    }
#pragma unroll
    for (int nt = 0; nt < 2; ++nt) {
      FragU fb;  // B(kdim, n) = K[n][kdim] -> row-major over kdim
      const bf16* bp = &Ks[(n0 + nt * 16 + l15) * ALP + ks];
#pragma unroll
      for (int d = 0; d < 8; ++d)
        fb.u[d] = *(const unsigned int*)(bp + (d < 4 ? 2 * d : 16 + 2 * (d - 4)) + khalf);
      sacc[nt] = wmma_bf16f32(fa.v, fb.v, sacc[nt]);
    }
  }
#pragma unroll
  for (int nt = 0; nt < 2; ++nt)
#pragma unroll
    for (int r = 0; r < 8; ++r)
      Sc[(m0 + mh + r) * SLP + n0 + nt * 16 + l15] = sacc[nt][r] * 0.125f;
  __syncthreads();

  // ---- softmax with masking, one row per thread (rows 0..63) ----
  if (tid < 64) {
    const int t = tid;
    float mx = -3.0e38f;
    for (int kt = 0; kt < 64; ++kt) {
      const bool msk = (kt >= Tk) || (causal && kt > t) ||
                       (ids && ids[b * 64 + kt] == 0) ||
                       (kpad && kpad[b * 64 + kt] != 0);
      const float v = msk ? -1.0e9f : Sc[t * SLP + kt];
      Sc[t * SLP + kt] = v;
      mx = fmaxf(mx, v);
    }
    float sum = 0.f;
    for (int kt = 0; kt < 64; ++kt) {
      const float e = __expf(Sc[t * SLP + kt] - mx);
      Sc[t * SLP + kt] = e;
      sum += e;
    }
    const float rinv = 1.f / (sum + 1e-20f);
    for (int kt = 0; kt < 64; ++kt)
      Qs[t * ALP + kt] = (bf16)(Sc[t * SLP + kt] * rinv);  // reuse Q tile
  }
  __syncthreads();

  // ---- phase B: O = attn @ V  (K dim = Tk padded to 64) ----
  v8f oacc[2];
#pragma unroll
  for (int nt = 0; nt < 2; ++nt)
#pragma unroll
    for (int r = 0; r < 8; ++r) oacc[nt][r] = 0.f;

  for (int ks = 0; ks < 64; ks += 32) {
    FragU fa;
    const bf16* ap = &Qs[(m0 + l15) * ALP + ks];
#pragma unroll
    for (int d = 0; d < 8; ++d) {
      const int klo = (d < 4 ? 2 * d : 16 + 2 * (d - 4)) + khalf;
      fa.u[d] = *(const unsigned int*)(ap + klo);
    }
#pragma unroll
    for (int nt = 0; nt < 2; ++nt) {
      FragU fb;  // B(k, n) = V[k][n]: gather pairs across LDS rows, pack
      const int n = n0 + nt * 16 + l15;
#pragma unroll
      for (int d = 0; d < 8; ++d) {
        const int klo = ks + (d < 4 ? 2 * d : 16 + 2 * (d - 4)) + khalf;
        const unsigned int lo = bf16bits(Vs[klo * ALP + n]);
        const unsigned int hi = bf16bits(Vs[(klo + 1) * ALP + n]);
        fb.u[d] = lo | (hi << 16);
      }
      oacc[nt] = wmma_bf16f32(fa.v, fb.v, oacc[nt]);
    }
  }
#pragma unroll
  for (int nt = 0; nt < 2; ++nt)
#pragma unroll
    for (int r = 0; r < 8; ++r) {
      const int m = m0 + mh + r;
      if (m < 63)
        O[(size_t)(b * 63 + m) * 512 + h * 64 + n0 + nt * 16 + l15] =
            (bf16)oacc[nt][r];
    }
}

// ---------------------------------------------------------------------------
// Host: full decoder pipeline.  Scratch carved from d_ws (bump allocator).
// Input order (setup_inputs dict order, layers dict in declaration order):
//  0 memory  1 seq_len  2 input_ids  3 memory_key_padding_mask
//  4 embedding  5 pos_embed
//  6 sa_in_w 7 sa_in_b 8 sa_out_w 9 sa_out_b 10 ca_in_w 11 ca_in_b
//  12 ca_out_w 13 ca_out_b 14 w1 15 b1 16 w2 17 b2
//  18 ln1_g 19 ln1_b 20 ln2_g 21 ln2_b 22 ln3_g 23 ln3_b
//  24 Wv 25 bv
// ---------------------------------------------------------------------------
extern "C" void kernel_launch(void* const* d_in, const int* in_sizes, int n_in,
                              void* d_out, int out_size, void* d_ws, size_t ws_size,
                              hipStream_t stream) {
  (void)in_sizes; (void)n_in; (void)out_size; (void)ws_size;
  constexpr int Bc = 32, T = 63, M = 2016, D = 512, FF = 2048, V = 32000,
                SM = 64, NL = 2;

  const float* memory   = (const float*)d_in[0];
  const int*   ids      = (const int*)d_in[2];
  const unsigned char* mkpm = (const unsigned char*)d_in[3];
  const float* emb      = (const float*)d_in[4];
  const float* pos      = (const float*)d_in[5];
  const float* sa_in_w  = (const float*)d_in[6];
  const float* sa_in_b  = (const float*)d_in[7];
  const float* sa_out_w = (const float*)d_in[8];
  const float* sa_out_b = (const float*)d_in[9];
  const float* ca_in_w  = (const float*)d_in[10];
  const float* ca_in_b  = (const float*)d_in[11];
  const float* ca_out_w = (const float*)d_in[12];
  const float* ca_out_b = (const float*)d_in[13];
  const float* w1       = (const float*)d_in[14];
  const float* b1       = (const float*)d_in[15];
  const float* w2       = (const float*)d_in[16];
  const float* b2       = (const float*)d_in[17];
  const float* ln1_g    = (const float*)d_in[18];
  const float* ln1_b    = (const float*)d_in[19];
  const float* ln2_g    = (const float*)d_in[20];
  const float* ln2_b    = (const float*)d_in[21];
  const float* ln3_g    = (const float*)d_in[22];
  const float* ln3_b    = (const float*)d_in[23];
  const float* Wv       = (const float*)d_in[24];
  const float* bv       = (const float*)d_in[25];

  char* ws = (char*)d_ws;
  size_t off = 0;
  auto alloc = [&](size_t bytes) -> void* {
    void* p = ws + off;
    off = (off + bytes + 255) & ~(size_t)255;
    return p;
  };

  bf16* sa_in_wb  = (bf16*)alloc((size_t)NL * 3 * D * D * 2);
  bf16* sa_out_wb = (bf16*)alloc((size_t)NL * D * D * 2);
  bf16* ca_in_wb  = (bf16*)alloc((size_t)NL * 3 * D * D * 2);
  bf16* ca_out_wb = (bf16*)alloc((size_t)NL * D * D * 2);
  bf16* w1b  = (bf16*)alloc((size_t)NL * FF * D * 2);
  bf16* w2b  = (bf16*)alloc((size_t)NL * D * FF * 2);
  bf16* Wvb  = (bf16*)alloc((size_t)V * D * 2);
  bf16* memB = (bf16*)alloc((size_t)Bc * SM * D * 2);
  float* x   = (float*)alloc((size_t)M * D * 4);
  bf16* xb   = (bf16*)alloc((size_t)M * D * 2);
  float* qkv = (float*)alloc((size_t)M * 3 * D * 4);
  float* qbf = (float*)alloc((size_t)M * D * 4);
  float* kvb = (float*)alloc((size_t)Bc * SM * 2 * D * 4);
  bf16* ob   = (bf16*)alloc((size_t)M * D * 2);
  float* t1  = (float*)alloc((size_t)M * D * 4);
  bf16* h1b  = (bf16*)alloc((size_t)M * FF * 2);

  auto conv = [&](const float* s, bf16* dst, long long n) {
    int blocks = (int)((n + 2047) / 2048);
    if (blocks > 4096) blocks = 4096;
    if (blocks < 1) blocks = 1;
    f32_to_bf16_kernel<<<blocks, 256, 0, stream>>>(s, dst, n);
  };
  conv(sa_in_w,  sa_in_wb,  (long long)NL * 3 * D * D);
  conv(sa_out_w, sa_out_wb, (long long)NL * D * D);
  conv(ca_in_w,  ca_in_wb,  (long long)NL * 3 * D * D);
  conv(ca_out_w, ca_out_wb, (long long)NL * D * D);
  conv(w1, w1b, (long long)NL * FF * D);
  conv(w2, w2b, (long long)NL * D * FF);
  conv(Wv, Wvb, (long long)V * D);
  conv(memory, memB, (long long)Bc * SM * D);

  embed_kernel<<<M, 256, 0, stream>>>(ids, emb, pos, x, xb);

  auto gemm = [&](const bf16* A, const bf16* W_, const float* bias,
                  float* Cf, bf16* Cb, int Mm, int Nn, int Kk, int relu) {
    dim3 grid((Nn + GBN - 1) / GBN, (Mm + GBM - 1) / GBM);
    gemm_bf16_kernel<<<grid, 256, 4 * GTILE_BYTES, stream>>>(
        A, W_, bias, Cf, Cb, Mm, Nn, Kk, relu);
  };

  for (int l = 0; l < NL; ++l) {
    const bf16* saw = sa_in_wb  + (size_t)l * 3 * D * D;
    const bf16* sow = sa_out_wb + (size_t)l * D * D;
    const bf16* caw = ca_in_wb  + (size_t)l * 3 * D * D;
    const bf16* cow = ca_out_wb + (size_t)l * D * D;
    const bf16* W1l = w1b + (size_t)l * FF * D;
    const bf16* W2l = w2b + (size_t)l * D * FF;

    // ---- self-attention block ----
    gemm(xb, saw, sa_in_b + l * 3 * D, qkv, nullptr, M, 3 * D, D, 0);
    attention_kernel<<<Bc * 8, 256, 0, stream>>>(
        qkv, 3 * D, 0, qkv, 3 * D, D, qkv, 2 * D, T, /*causal=*/1, ids, nullptr,
        ob);
    gemm(ob, sow, sa_out_b + l * D, t1, nullptr, M, D, D, 0);
    add_ln_kernel<<<M, 256, 0, stream>>>(x, t1, ln1_g + l * D, ln1_b + l * D, x,
                                         xb);

    // ---- cross-attention block ----
    gemm(xb, caw, ca_in_b + l * 3 * D, qbf, nullptr, M, D, D, 0);
    gemm(memB, caw + (size_t)D * D, ca_in_b + l * 3 * D + D, kvb, nullptr,
         Bc * SM, 2 * D, D, 0);
    attention_kernel<<<Bc * 8, 256, 0, stream>>>(
        qbf, D, 0, kvb, 2 * D, 0, kvb, D, SM, /*causal=*/0, nullptr, mkpm, ob);
    gemm(ob, cow, ca_out_b + l * D, t1, nullptr, M, D, D, 0);
    add_ln_kernel<<<M, 256, 0, stream>>>(x, t1, ln2_g + l * D, ln2_b + l * D, x,
                                         xb);

    // ---- FFN block (ReLU fused into first GEMM, bf16 out) ----
    gemm(xb, W1l, b1 + l * FF, nullptr, h1b, M, FF, D, /*relu=*/1);
    gemm(h1b, W2l, b2 + l * D, t1, nullptr, M, D, FF, 0);
    add_ln_kernel<<<M, 256, 0, stream>>>(x, t1, ln3_g + l * D, ln3_b + l * D, x,
                                         xb);
  }

  // ---- vocab projection straight into d_out (f32, 2016 x 32000) ----
  gemm(xb, Wvb, bv, (float*)d_out, nullptr, M, V, D, 0);
}
